// VectorQuantizer_13494787244639
// MI455X (gfx1250) — matrix operand order repaired and checked by reading, not compiled, not run on hardware
//
#include <hip/hip_runtime.h>
#include <hip/hip_bf16.h>

typedef __attribute__((ext_vector_type(16))) __bf16 v16bf;
typedef __attribute__((ext_vector_type(8)))  float  v8f;
typedef __attribute__((ext_vector_type(4)))  unsigned int v4u;
typedef __attribute__((ext_vector_type(8)))  int    v8i;
typedef __attribute__((ext_vector_type(4)))  int    v4i;

#define D_DIM 64

#if defined(__has_builtin)
#if __has_builtin(__builtin_amdgcn_tensor_load_to_lds) && \
    __has_builtin(__builtin_amdgcn_s_wait_tensorcnt)
#define VQ_USE_TDM 1
#endif
#endif

// ---------------------------------------------------------------------------
// Prep: transpose codebook [D=64][K] -> [K][64] in fp32 (exact gather) and
// bf16 (WMMA B operands), compute per-code squared norms, zero loss slot.
// ---------------------------------------------------------------------------
__global__ void vq_prep_kernel(const float* __restrict__ emb, int K,
                               float* __restrict__ codeT_f32,
                               __bf16* __restrict__ codeT_bf16,
                               float* __restrict__ enorm,
                               float* __restrict__ loss_out) {
  int k = blockIdx.x * blockDim.x + threadIdx.x;
  if (k == 0) *loss_out = 0.0f;
  if (k >= K) return;
  float s = 0.0f;
  for (int d = 0; d < D_DIM; ++d) {
    float v = emb[d * K + k];       // coalesced: consecutive k adjacent
    codeT_f32[k * D_DIM + d] = v;
    codeT_bf16[k * D_DIM + d] = (__bf16)v;
    s += v * v;
  }
  enorm[k] = s;
}

// ---------------------------------------------------------------------------
// Main: per-wave 16-row tile; distances via v_wmma_f32_16x16x32_bf16;
// codebook staged into LDS per 64KB phase via the Tensor Data Mover;
// software-pipelined B-fragment loads; argmin via shfl butterflies;
// fp32 gather + loss accumulation.
// ---------------------------------------------------------------------------
__global__ void __launch_bounds__(256) vq_main_kernel(
    const float* __restrict__ x,        // [rows][64]
    const __bf16* __restrict__ cb_bf,   // [K][64]
    const float* __restrict__ cb_f32,   // [K][64]
    const float* __restrict__ enorm_g,  // [K]
    float* __restrict__ out,            // [rows][64]
    float* __restrict__ loss,
    int K, int rows) {
  extern __shared__ unsigned char smem[];   // (K/2)*64 bf16 = 64 KB for K=1024
  __bf16* cb = (__bf16*)smem;

  const int tid  = threadIdx.x;
  const int lane = tid & 31;
  const int wave = tid >> 5;
  const int h    = lane >> 4;   // half-wave select
  const int m    = lane & 15;   // A: row-in-tile; B/C: column (code) index

  const int rt = blockIdx.x * 128 + wave * 16;   // 8 waves * 16 rows = 128/block
  const float* xr = x + (size_t)(rt + m) * D_DIM;

  // A fragments in contiguous k order (same as ldB):
  // lane (m, h): elems 0..15 = K s+16h+{0..15}
  auto make_frag = [&](int s) -> v16bf {
    v16bf a;
    const float4 u0 = *(const float4*)(xr + s + 16 * h);
    const float4 u1 = *(const float4*)(xr + s + 16 * h + 4);
    const float4 u2 = *(const float4*)(xr + s + 16 * h + 8);
    const float4 u3 = *(const float4*)(xr + s + 16 * h + 12);
    a[0]  = (__bf16)u0.x; a[1]  = (__bf16)u0.y; a[2]  = (__bf16)u0.z; a[3]  = (__bf16)u0.w;
    a[4]  = (__bf16)u1.x; a[5]  = (__bf16)u1.y; a[6]  = (__bf16)u1.z; a[7]  = (__bf16)u1.w;
    a[8]  = (__bf16)u2.x; a[9]  = (__bf16)u2.y; a[10] = (__bf16)u2.z; a[11] = (__bf16)u2.w;
    a[12] = (__bf16)u3.x; a[13] = (__bf16)u3.y; a[14] = (__bf16)u3.z; a[15] = (__bf16)u3.w;
    return a;
  };
  const v16bf a0 = make_frag(0);
  const v16bf a1 = make_frag(32);

  // B fragment loader, contiguous k order (as A): lane = column; elems = 16 K values,
  // halves split K{0..15}/{16..31} within the 32-wide k-step. 32B aligned.
  auto ldB = [&](int t, int s) -> v16bf {
    return *(const v16bf*)(cb + (t * 16 + m) * D_DIM + s + 16 * h);
  };

  float best_d[8];
  int   best_i[8];
#pragma unroll
  for (int r = 0; r < 8; ++r) { best_d[r] = 3.402823466e38f; best_i[r] = 0; }

  const int KH = K / 2;                 // codes per LDS phase (512)
  const int NT = KH / 16;               // n-tiles per phase (32)

  for (int kh = 0; kh < 2; ++kh) {
    __syncthreads();                    // previous-phase readers done
#ifdef VQ_USE_TDM
    if (wave == 0) {
      // D# per CDNA5 ISA: 2D tensor = [KH rows][64 cols] of 2-byte elements,
      // tile == whole tensor, contiguous rows (stride 64), dest LDS offset 0.
      const unsigned long long ga =
          (unsigned long long)(cb_bf + (size_t)kh * KH * D_DIM);
      v4u g0;
      g0[0] = 1u;                                   // count=1 valid user D#
      g0[1] = 0u;                                   // lds_addr = 0
      g0[2] = (unsigned)(ga & 0xffffffffu);         // global_addr[31:0]
      g0[3] = (unsigned)((ga >> 32) & 0x01ffffffu)  // global_addr[56:32]
              | (2u << 30);                         // type = 2 (image)
      v8i g1;
      g1[0] = (int)(1u << 16);                      // data_size=1 (2 bytes)
      g1[1] = (int)((unsigned)D_DIM << 16);         // tensor_dim0 = 64 (lo16)
      g1[2] = (int)(((unsigned)KH & 0xffffu) << 16);// dim0 hi=0 | tensor_dim1
      g1[3] = (int)((unsigned)D_DIM << 16);         // dim1 hi=0 | tile_dim0=64
      g1[4] = (int)((unsigned)KH & 0xffffu);        // tile_dim1=KH, tile_dim2=0
      g1[5] = D_DIM;                                // tensor_dim0_stride lo32
      g1[6] = 0;                                    // stride hi, dim1_stride lo
      g1[7] = 0;
      v4i gz = {0, 0, 0, 0};
#if defined(__clang_major__) && (__clang_major__ >= 23)
      v8i gz8 = {0, 0, 0, 0, 0, 0, 0, 0};
      __builtin_amdgcn_tensor_load_to_lds(g0, g1, gz, gz, gz8, 0);
#else
      __builtin_amdgcn_tensor_load_to_lds(g0, g1, gz, gz, 0);
#endif
      __builtin_amdgcn_s_wait_tensorcnt(0);
    }
#else
    {
      const int chunks = KH * D_DIM * 2 / 16;
      const uint4* src = (const uint4*)(cb_bf + (size_t)kh * KH * D_DIM);
      uint4* dst = (uint4*)cb;
      for (int i = tid; i < chunks; i += 256) dst[i] = src[i];
    }
#endif
    __syncthreads();                    // LDS tile ready for all waves

    // Software-pipelined n-tile loop: fragments/norm for tile t+1 are loaded
    // while tile t feeds the WMMA + argmin, so the wait becomes dscnt<=4
    // instead of dscnt==0 (LDS latency overlaps the matrix pipe).
    v16bf b0 = ldB(0, 0);
    v16bf b1 = ldB(0, 32);
    float e2 = enorm_g[kh * KH + m];

#pragma unroll 2
    for (int t = 0; t < NT; ++t) {
      const int tn = (t + 1 < NT) ? (t + 1) : t;   // clamp: last iter reloads
      const v16bf nb0 = ldB(tn, 0);
      const v16bf nb1 = ldB(tn, 32);
      const float ne2 = enorm_g[kh * KH + tn * 16 + m];

      v8f acc = {};
      acc = __builtin_amdgcn_wmma_f32_16x16x32_bf16(false, a0, false, b0,
                                                    (short)0, acc, false, false);
      acc = __builtin_amdgcn_wmma_f32_16x16x32_bf16(false, a1, false, b1,
                                                    (short)0, acc, false, false);
      // argmin over ||e||^2 - 2*x.e  (||x||^2 is row-constant; irrelevant)
      const int code = kh * KH + t * 16 + m;
#pragma unroll
      for (int r = 0; r < 8; ++r) {
        const float dd = fmaf(-2.0f, acc[r], e2);
        if (dd < best_d[r]) { best_d[r] = dd; best_i[r] = code; }
      }
      b0 = nb0;
      b1 = nb1;
      e2 = ne2;
    }
  }

  // C layout: VGPR r holds row r (lanes 0-15) / row r+8 (lanes 16-31), with
  // the 16 codes of a tile spread across the 16 lanes of each half.
  float lsum = 0.0f;
#pragma unroll
  for (int r = 0; r < 8; ++r) {
    float bd = best_d[r];
    int   bi = best_i[r];
#pragma unroll
    for (int mask = 8; mask >= 1; mask >>= 1) {   // butterfly within each half
      const float od = __shfl_xor(bd, mask, 32);
      const int   oi = __shfl_xor(bi, mask, 32);
      if (od < bd || (od == bd && oi < bi)) { bd = od; bi = oi; }
    }
    const int row = rt + r + 8 * h;
    const float4 q  = *(const float4*)(cb_f32 + (size_t)bi * D_DIM + 4 * m);
    const float4 xv = *(const float4*)(x + (size_t)row * D_DIM + 4 * m);
    *(float4*)(out + (size_t)row * D_DIM + 4 * m) = q;
    const float dx = q.x - xv.x, dy = q.y - xv.y;
    const float dz = q.z - xv.z, dw = q.w - xv.w;
    lsum += dx * dx + dy * dy + dz * dz + dw * dw;
  }

  // Per-wave loss reduction, one atomic per wave.
#pragma unroll
  for (int mask = 16; mask >= 1; mask >>= 1) lsum += __shfl_xor(lsum, mask, 32);
  if (lane == 0) {
    const float scale = 1.25f / ((float)rows * (float)D_DIM);
    atomicAdd(loss, lsum * scale);
  }
}

// ---------------------------------------------------------------------------
extern "C" void kernel_launch(void* const* d_in, const int* in_sizes, int n_in,
                              void* d_out, int out_size, void* d_ws, size_t ws_size,
                              hipStream_t stream) {
  const float* x   = (const float*)d_in[0];
  const float* emb = (const float*)d_in[1];
  const int NE   = in_sizes[0];        // 4194304
  const int rows = NE / D_DIM;         // 65536
  const int K    = in_sizes[1] / D_DIM;// 1024

  float* out  = (float*)d_out;
  float* loss = out + NE;              // scalar loss appended after quantized

  // Workspace layout: fp32 codeT [K][64] | bf16 codeT [K][64] | enorm [K]
  float*  codeT_f32 = (float*)d_ws;
  __bf16* codeT_bf  = (__bf16*)(codeT_f32 + (size_t)K * D_DIM);
  float*  enorm     = (float*)((char*)codeT_bf + (size_t)K * D_DIM * sizeof(__bf16));

  vq_prep_kernel<<<(K + 255) / 256, 256, 0, stream>>>(emb, K, codeT_f32,
                                                      codeT_bf, enorm, loss);

  const size_t smem = (size_t)(K / 2) * D_DIM * sizeof(__bf16);  // 64 KB
  const int blocks = rows / 128;                                  // 512
  vq_main_kernel<<<blocks, 256, smem, stream>>>(x, codeT_bf, codeT_f32, enorm,
                                                out, loss, K, rows);
}
